// EdgeConv2_85495618994322
// MI455X (gfx1250) — matrix operand order repaired and checked by reading, not compiled, not run on hardware
//
#include <hip/hip_runtime.h>
#include <hip/hip_bf16.h>
#include <math.h>
#include <stdint.h>

#define B_    8
#define NPTS  4096
#define KNB   20
#define K2NB  40
#define COLS  (NPTS * KNB)   // 81920 columns per batch in the edge-conv GEMMs
#define GEPS  1e-5f

typedef __attribute__((ext_vector_type(16))) _Float16 v16h;
typedef __attribute__((ext_vector_type(8)))  _Float16 h8;
typedef __attribute__((ext_vector_type(8)))  float    v8f;
typedef unsigned int u32;
typedef __attribute__((ext_vector_type(4))) u32 u32x4;
typedef __attribute__((ext_vector_type(8))) u32 u32x8;

// Fixed stand-in for jax.random.permutation(key(42), 40)[:20] (not reproducible
// without the Threefry stream; 20 distinct indices in [0,40)).
__constant__ int c_sel[KNB] = {12,33,7,21,0,38,15,26,3,19,31,8,24,5,36,11,28,17,2,39};

__global__ void zero_f32(float* p, int n) {
    int i = blockIdx.x * blockDim.x + threadIdx.x;
    if (i < n) p[i] = 0.f;
}

// ---------------------------------------------------------------------------
// Tensor Data Mover: 1-D tile copy global -> LDS, nelem2 2-byte elements.
// D# per CDNA5 ISA ch.8: group0 = {count|flags, lds_addr, global_addr, type=2},
// group1 = {mask/data_size, tensor_dim0/1, tile_dim0/1/2, strides}.
// Issued once by one wave; completion via s_wait_tensorcnt.
// ---------------------------------------------------------------------------
__device__ __forceinline__ void tdm_load_to_lds_1d(const void* gptr, u32 lds_byte_off, u32 nelem2) {
    unsigned long long ga = (unsigned long long)(uintptr_t)gptr;
    u32x4 g0;
    g0[0] = 1u;                                             // count=1 (valid user desc)
    g0[1] = lds_byte_off;                                   // lds_addr
    g0[2] = (u32)(ga & 0xffffffffu);                        // global_addr[31:0]
    g0[3] = (u32)((ga >> 32) & 0x01ffffffu) | (2u << 30);   // global_addr[56:32] | type=2
    u32x8 g1;
    g1[0] = (1u << 16);                                     // workgroup_mask=0, data_size=1 (2B)
    g1[1] = (nelem2 & 0xffffu) << 16;                       // tensor_dim0[15:0]
    g1[2] = ((nelem2 >> 16) & 0xffffu) | (1u << 16);        // tensor_dim0[31:16], tensor_dim1=1
    g1[3] = (nelem2 & 0xffffu) << 16;                       // tile_dim0 = nelem2
    g1[4] = 1u;                                             // tile_dim1=1, tile_dim2=0
    g1[5] = nelem2;                                         // tensor_dim0_stride[31:0]
    g1[6] = 0u;                                             // stride hi / dim1_stride lo
    g1[7] = 0u;
    asm volatile("tensor_load_to_lds %0, %1" :: "s"(g0), "s"(g1) : "memory");
}

// ---------------------------------------------------------------------------
// kNN: one block per (b,n). All 4096 distances in LDS, then 40 tie-broken
// argmax passes -> sorted top-40 neighbor indices (top-20 = first 20).
// ---------------------------------------------------------------------------
__global__ void knn_topk(const float* __restrict__ x, int* __restrict__ idx40) {
    const int bn = blockIdx.x;
    const int b  = bn / NPTS;
    const int n  = bn % NPTS;
    __shared__ float dist[NPTS];
    __shared__ float rv[128];
    __shared__ int   ri[128];
    const int t = threadIdx.x;  // 128 threads
    const float* xb = x + (size_t)b * 3 * NPTS;
    const float q0 = xb[n], q1 = xb[NPTS + n], q2 = xb[2 * NPTS + n];
    const float qq = q0 * q0 + q1 * q1 + q2 * q2;
    for (int m = t; m < NPTS; m += 128) {
        float m0 = xb[m], m1 = xb[NPTS + m], m2 = xb[2 * NPTS + m];
        float inner = q0 * m0 + q1 * m1 + q2 * m2;
        float mm = m0 * m0 + m1 * m1 + m2 * m2;
        dist[m] = 2.f * inner - qq - mm;   // pd (<= 0, max == closest)
    }
    __syncthreads();
    int* outp = idx40 + (size_t)bn * K2NB;
    for (int it = 0; it < K2NB; ++it) {
        float best = -3.4e38f; int bi = 0x7fffffff;
        for (int m = t; m < NPTS; m += 128) {
            float d = dist[m];
            if (d > best || (d == best && m < bi)) { best = d; bi = m; }
        }
        rv[t] = best; ri[t] = bi;
        __syncthreads();
        for (int s = 64; s > 0; s >>= 1) {
            if (t < s) {
                float ov = rv[t + s]; int oi = ri[t + s];
                if (ov > rv[t] || (ov == rv[t] && oi < ri[t])) { rv[t] = ov; ri[t] = oi; }
            }
            __syncthreads();
        }
        if (t == 0) { outp[it] = ri[0]; dist[ri[0]] = -3.4e38f; }
        __syncthreads();
    }
}

// ---------------------------------------------------------------------------
// Gather edge features + 6->64 conv, store f16 channel-major:
// y1[(b*64 + c)*COLS + n*20 + kk]  -> contiguous 20-halfword runs per thread,
// and contiguous 16-column vector loads for the WMMA B operand downstream.
// ---------------------------------------------------------------------------
__global__ void gather_conv1(const float* __restrict__ x, const int* __restrict__ idx40,
                             const float* __restrict__ W1, _Float16* __restrict__ y1,
                             float* __restrict__ gn_accum, int useSel) {
    const int bn = blockIdx.x;
    const int b  = bn / NPTS;
    const int n  = bn % NPTS;
    __shared__ float f1[6][KNB];
    __shared__ float sSum[32], sSqs[32];
    const int t = threadIdx.x;  // 64 threads
    if (t < 32) { sSum[t] = 0.f; sSqs[t] = 0.f; }
    const float* xb = x + (size_t)b * 3 * NPTS;
    if (t < KNB) {
        int src = useSel ? c_sel[t] : t;
        int nb  = idx40[(size_t)bn * K2NB + src];
        #pragma unroll
        for (int c = 0; c < 3; ++c) {
            float fc = xb[c * NPTS + nb];
            float cc = xb[c * NPTS + n];
            f1[c][t]     = fc - cc;   // feat - center
            f1[3 + c][t] = cc;        // center
        }
    }
    __syncthreads();
    const int c = t;
    float w[6];
    #pragma unroll
    for (int i = 0; i < 6; ++i) w[i] = W1[c * 6 + i];
    float lsum = 0.f, lsq = 0.f;
    size_t base = ((size_t)b * 64 + c) * (size_t)COLS + (size_t)n * KNB;
    for (int kk = 0; kk < KNB; ++kk) {
        float v = 0.f;
        #pragma unroll
        for (int i = 0; i < 6; ++i) v += w[i] * f1[i][kk];
        y1[base + kk] = (_Float16)v;
        lsum += v; lsq += v * v;
    }
    atomicAdd(&sSum[c >> 1], lsum);
    atomicAdd(&sSqs[c >> 1], lsq);
    __syncthreads();
    if (t < 32) {
        atomicAdd(&gn_accum[(b * 32 + t) * 2 + 0], sSum[t]);
        atomicAdd(&gn_accum[(b * 32 + t) * 2 + 1], sSqs[t]);
    }
}

// In-place: [sum,sumsq] per (b,group) -> [mean, inv_std]
__global__ void finalize_gn(float* st, float count) {
    int i = threadIdx.x;  // 256 = 8 batches * 32 groups
    if (i < 256) {
        float s = st[i * 2], q = st[i * 2 + 1];
        float mean = s / count;
        float var  = q / count - mean * mean;
        st[i * 2]     = mean;
        st[i * 2 + 1] = rsqrtf(var + GEPS);
    }
}

__device__ __forceinline__ float lrelu(float v) { return v >= 0.f ? v : 0.2f * v; }

// ---------------------------------------------------------------------------
// 64->64 conv over COLS columns/batch via v_wmma_f32_16x16x32_f16.
// Weights staged global->LDS by the Tensor Data Mover, then converted to f16.
// GN(layer1)+lrelu fused into the B-operand load; GN stats for layer2 output.
// y2 stays column-major (channel-contiguous) so D stores pack into b128.
// ---------------------------------------------------------------------------
__global__ void conv2_wmma(const _Float16* __restrict__ y1, _Float16* __restrict__ y2,
                           const float* __restrict__ W2, const float* __restrict__ gamma,
                           const float* __restrict__ beta, const float* __restrict__ gn_in,
                           float* __restrict__ gn_out) {
    const int b = blockIdx.y;
    __shared__ float    sWf[64 * 64];
    __shared__ _Float16 sW[64 * 64];
    __shared__ float sScale[64], sShift[64];
    __shared__ float sSum[32], sSqs[32];
    const int tid = threadIdx.x;  // 256 = 8 waves

    if (tid == 0) {   // TDM: 64x64 f32 weights = 8192 two-byte elements * 2
        tdm_load_to_lds_1d(W2, (u32)(uintptr_t)sWf, 64 * 64 * 2);
        __builtin_amdgcn_s_wait_tensorcnt(0);
    }
    if (tid < 64) {
        int g = tid >> 1;
        float mean = gn_in[(b * 32 + g) * 2 + 0];
        float inv  = gn_in[(b * 32 + g) * 2 + 1];
        sScale[tid] = gamma[tid] * inv;
        sShift[tid] = beta[tid] - mean * inv * gamma[tid];
    }
    if (tid < 32) { sSum[tid] = 0.f; sSqs[tid] = 0.f; }
    __syncthreads();
    for (int i = tid; i < 64 * 64; i += 256) sW[i] = (_Float16)sWf[i];
    __syncthreads();

    const int wave = tid >> 5;
    const int lane = tid & 31;
    const int col0 = (blockIdx.x * 8 + wave) * 16;   // within [0, COLS)

    // B operand: lane%16 = column, element = K-row (input channel) in ISA table order.
    v16h bfrag[2];
    #pragma unroll
    for (int ks = 0; ks < 2; ++ks) {
        #pragma unroll
        for (int e = 0; e < 16; ++e) {
            int k = (e < 8) ? (e + 8 * (lane >> 4)) : (e + 8 + 8 * (lane >> 4));
            int ch = ks * 32 + k;
            float sc = sScale[ch], sh = sShift[ch];
            float v = (float)y1[((size_t)b * 64 + ch) * (size_t)COLS + col0 + (lane & 15)];
            bfrag[ks][e] = (_Float16)lrelu(v * sc + sh);
        }
    }

    const int half = lane >> 4;
    const int mloc = lane & 15;
    const int coln = col0 + (lane & 15);
    const size_t obase = (size_t)b * COLS;
    #pragma unroll
    for (int mt = 0; mt < 4; ++mt) {
        v16h a0{}, a1{};
        #pragma unroll
        for (int e = 0; e < 16; ++e) {
            int k = (e < 8) ? (e + 8 * half) : (e + 8 + 8 * half);  // ISA A 16x32 layout
            a0[e] = sW[(mt * 16 + mloc) * 64 + k];
            a1[e] = sW[(mt * 16 + mloc) * 64 + 32 + k];
        }
        v8f acc{};
        acc = __builtin_amdgcn_wmma_f32_16x16x32_f16(false, a0, false, bfrag[0], (short)0, acc, false, false);
        acc = __builtin_amdgcn_wmma_f32_16x16x32_f16(false, a1, false, bfrag[1], (short)0, acc, false, false);
        #pragma unroll
        for (int r = 0; r < 8; ++r) {
            int oc = mt * 16 + r + 8 * half;    // ISA C/D layout
            y2[(obase + coln) * 64 + oc] = (_Float16)acc[r];
        }
        #pragma unroll
        for (int r = 0; r < 8; r += 2) {        // r-pairs share a GN group
            int g = (mt * 16 + r + 8 * half) >> 1;
            float v0 = acc[r], v1 = acc[r + 1];
            atomicAdd(&sSum[g], v0 + v1);
            atomicAdd(&sSqs[g], v0 * v0 + v1 * v1);
        }
    }
    __syncthreads();
    if (tid < 32) {
        atomicAdd(&gn_out[(b * 32 + tid) * 2 + 0], sSum[tid]);
        atomicAdd(&gn_out[(b * 32 + tid) * 2 + 1], sSqs[tid]);
    }
}

// GN(layer2)+lrelu fused with max/mean pooling over the 20 neighbors.
// Output x1raw: (B,128,N) = [max ; mean].
__global__ void pool_norm(const _Float16* __restrict__ y2, const float* __restrict__ gamma,
                          const float* __restrict__ beta, const float* __restrict__ gn,
                          float* __restrict__ x1raw) {
    const int bn = blockIdx.x;
    const int b  = bn / NPTS;
    const int n  = bn % NPTS;
    const int c  = threadIdx.x;  // 64
    const int g  = c >> 1;
    float mean = gn[(b * 32 + g) * 2 + 0];
    float inv  = gn[(b * 32 + g) * 2 + 1];
    float sc = gamma[c] * inv;
    float sh = beta[c] - mean * inv * gamma[c];
    float mx = -3.4e38f, sm = 0.f;
    size_t base = (size_t)bn * KNB * 64 + c;
    for (int kk = 0; kk < KNB; ++kk) {
        float v = lrelu((float)y2[base + (size_t)kk * 64] * sc + sh);
        mx = fmaxf(mx, v); sm += v;
    }
    x1raw[((size_t)b * 128 + c) * NPTS + n]      = mx;
    x1raw[((size_t)b * 128 + 64 + c) * NPTS + n] = sm * (1.f / (float)KNB);
}

// ---------------------------------------------------------------------------
// 128->64 conv1d via WMMA: M=64 (4 tiles), K=128 (4 steps), 16 cols per wave.
// Weights TDM-staged; writes raw conv output (f32) + accumulates GN stats.
// ---------------------------------------------------------------------------
__global__ void conv1d_wmma(const float* __restrict__ X, const float* __restrict__ W,
                            float* __restrict__ Yraw, float* __restrict__ gn_accum) {
    const int b = blockIdx.y;
    __shared__ float    sWf[64 * 128];
    __shared__ _Float16 sW[64 * 128];
    __shared__ float sSum[32], sSqs[32];
    const int tid = threadIdx.x;  // 256

    if (tid == 0) {   // TDM: 64x128 f32 = 16384 two-byte elements * 2
        tdm_load_to_lds_1d(W, (u32)(uintptr_t)sWf, 64 * 128 * 2);
        __builtin_amdgcn_s_wait_tensorcnt(0);
    }
    if (tid < 32) { sSum[tid] = 0.f; sSqs[tid] = 0.f; }
    __syncthreads();
    for (int i = tid; i < 64 * 128; i += 256) sW[i] = (_Float16)sWf[i];
    __syncthreads();

    const int wave = tid >> 5;
    const int lane = tid & 31;
    const int col0 = (blockIdx.x * 8 + wave) * 16;   // within [0, N)

    v16h bfrag[4];
    #pragma unroll
    for (int ks = 0; ks < 4; ++ks) {
        #pragma unroll
        for (int e = 0; e < 16; ++e) {
            int k = (e < 8) ? (e + 8 * (lane >> 4)) : (e + 8 + 8 * (lane >> 4));
            int ch = ks * 32 + k;
            bfrag[ks][e] = (_Float16)X[((size_t)b * 128 + ch) * NPTS + col0 + (lane & 15)];
        }
    }

    const int half = lane >> 4;
    const int mloc = lane & 15;
    const int n    = col0 + (lane & 15);
    #pragma unroll
    for (int mt = 0; mt < 4; ++mt) {
        v8f acc{};
        #pragma unroll
        for (int ks = 0; ks < 4; ++ks) {
            v16h a{};
            #pragma unroll
            for (int e = 0; e < 16; ++e) {
                int k = (e < 8) ? (e + 8 * half) : (e + 8 + 8 * half);
                a[e] = sW[(mt * 16 + mloc) * 128 + ks * 32 + k];
            }
            acc = __builtin_amdgcn_wmma_f32_16x16x32_f16(false, a, false, bfrag[ks], (short)0, acc, false, false);
        }
        #pragma unroll
        for (int r = 0; r < 8; ++r) {
            int oc = mt * 16 + r + 8 * half;
            Yraw[((size_t)b * 64 + oc) * NPTS + n] = acc[r];
        }
        #pragma unroll
        for (int r = 0; r < 8; r += 2) {
            int g = (mt * 16 + r + 8 * half) >> 1;
            float v0 = acc[r], v1 = acc[r + 1];
            atomicAdd(&sSum[g], v0 + v1);
            atomicAdd(&sSqs[g], v0 * v0 + v1 * v1);
        }
    }
    __syncthreads();
    if (tid < 32) {
        atomicAdd(&gn_accum[(b * 32 + tid) * 2 + 0], sSum[tid]);
        atomicAdd(&gn_accum[(b * 32 + tid) * 2 + 1], sSqs[tid]);
    }
}

// GN+lrelu over a (b,c) row; optional per-channel mean accumulation for SE.
__global__ void row_norm(const float* __restrict__ Yraw, const float* __restrict__ gamma,
                         const float* __restrict__ beta, const float* __restrict__ gn,
                         float* __restrict__ Out, float* __restrict__ chanSum, int doSE) {
    const int bc = blockIdx.x;  // B*64
    const int b = bc >> 6, c = bc & 63;
    const int g = c >> 1;
    float mean = gn[(b * 32 + g) * 2 + 0];
    float inv  = gn[(b * 32 + g) * 2 + 1];
    float sc = gamma[c] * inv;
    float sh = beta[c] - mean * inv * gamma[c];
    __shared__ float red[256];
    float local = 0.f;
    size_t base = ((size_t)b * 64 + c) * NPTS;
    for (int n = threadIdx.x; n < NPTS; n += 256) {
        float v = lrelu(Yraw[base + n] * sc + sh);
        Out[base + n] = v;
        local += v;
    }
    if (doSE) {
        red[threadIdx.x] = local;
        __syncthreads();
        for (int s = 128; s > 0; s >>= 1) {
            if (threadIdx.x < s) red[threadIdx.x] += red[threadIdx.x + s];
            __syncthreads();
        }
        if (threadIdx.x == 0) chanSum[b * 64 + c] = red[0];
    }
}

// Squeeze-excite gate: e = sigmoid(relu(s @ A^T) @ Bw^T), s = channel mean.
__global__ void se_excite(const float* __restrict__ chanSum, const float* __restrict__ A,
                          const float* __restrict__ Bw, float* __restrict__ e) {
    const int b = blockIdx.x;
    const int c = threadIdx.x;  // 64
    __shared__ float s[64], h[4];
    s[c] = chanSum[b * 64 + c] * (1.f / (float)NPTS);
    __syncthreads();
    if (c < 4) {
        float acc = 0.f;
        for (int i = 0; i < 64; ++i) acc += A[c * 64 + i] * s[i];
        h[c] = fmaxf(acc, 0.f);
    }
    __syncthreads();
    float acc = 0.f;
    #pragma unroll
    for (int j = 0; j < 4; ++j) acc += Bw[c * 4 + j] * h[j];
    e[b * 64 + c] = 1.f / (1.f + expf(-acc));
}

// Apply gate and place branch into its half of the concatenated (B,128,N).
__global__ void se_apply(const float* __restrict__ Xn, const float* __restrict__ e,
                         float* __restrict__ Xcat, int chanOff) {
    const int bc = blockIdx.x;
    const int b = bc >> 6, c = bc & 63;
    float ev = e[b * 64 + c];
    size_t src = ((size_t)b * 64 + c) * NPTS;
    size_t dst = ((size_t)b * 128 + chanOff + c) * NPTS;
    for (int n = threadIdx.x; n < NPTS; n += 256) Xcat[dst + n] = Xn[src + n] * ev;
}

// ---------------------------------------------------------------------------
extern "C" void kernel_launch(void* const* d_in, const int* in_sizes, int n_in,
                              void* d_out, int out_size, void* d_ws, size_t ws_size,
                              hipStream_t stream) {
    const float* x   = (const float*)d_in[0];
    const float* w1  = (const float*)d_in[1];
    const float* g1  = (const float*)d_in[2];
    const float* b1  = (const float*)d_in[3];
    const float* w2  = (const float*)d_in[4];
    const float* g2  = (const float*)d_in[5];
    const float* b2  = (const float*)d_in[6];
    const float* w12 = (const float*)d_in[7];
    const float* g12 = (const float*)d_in[8];
    const float* b12 = (const float*)d_in[9];
    const float* w22 = (const float*)d_in[10];
    const float* g22 = (const float*)d_in[11];
    const float* b22 = (const float*)d_in[12];
    const float* wc  = (const float*)d_in[13];
    const float* gc  = (const float*)d_in[14];
    const float* bc  = (const float*)d_in[15];
    const float* wc2 = (const float*)d_in[16];
    const float* gc2 = (const float*)d_in[17];
    const float* bc2 = (const float*)d_in[18];
    const float* wa  = (const float*)d_in[19];
    const float* ga  = (const float*)d_in[20];
    const float* ba  = (const float*)d_in[21];
    const float* s1a = (const float*)d_in[22];
    const float* s1b = (const float*)d_in[23];
    const float* s2a = (const float*)d_in[24];
    const float* s2b = (const float*)d_in[25];

    char* ws = (char*)d_ws;
    size_t off = 0;
    auto take = [&](size_t bytes) -> void* {
        void* p = ws + off;
        off = (off + bytes + 255) & ~(size_t)255;
        return p;
    };
    int*      idx40 = (int*)     take((size_t)B_ * NPTS * K2NB * sizeof(int));
    _Float16* y1    = (_Float16*)take((size_t)B_ * 64 * COLS * sizeof(_Float16));
    _Float16* y2    = (_Float16*)take((size_t)B_ * 64 * COLS * sizeof(_Float16));
    float*    x1raw = (float*)   take((size_t)B_ * 128 * NPTS * sizeof(float));
    float*    yraw  = (float*)   take((size_t)B_ * 64 * NPTS * sizeof(float));
    float*    x1n   = (float*)   take((size_t)B_ * 64 * NPTS * sizeof(float));
    float*    xcat  = (float*)   take((size_t)B_ * 128 * NPTS * sizeof(float));
    float*    stats = (float*)   take(8192 * sizeof(float));

    float* gnA1 = stats + 0 * 512;   // branch1 layer1
    float* gnA2 = stats + 1 * 512;   // branch1 layer2
    float* gnAc = stats + 2 * 512;   // branch1 conv1d
    float* gnB1 = stats + 3 * 512;   // branch2 layer1
    float* gnB2 = stats + 4 * 512;   // branch2 layer2
    float* gnBc = stats + 5 * 512;   // branch2 conv1d
    float* gnF  = stats + 6 * 512;   // final conv1d
    float* cs1  = stats + 7 * 512;   // SE channel sums, branch1
    float* cs2  = stats + 8 * 512;
    float* e1   = stats + 9 * 512;   // SE gates
    float* e2   = stats + 10 * 512;

    zero_f32<<<(7 * 512 + 255) / 256, 256, 0, stream>>>(stats, 7 * 512);
    knn_topk<<<B_ * NPTS, 128, 0, stream>>>(x, idx40);

    const float cntL = (float)(2 * NPTS * KNB);   // per-group count, layer GNs
    const float cntC = (float)(2 * NPTS);         // per-group count, conv1d GNs

    // ---- branch 1 (sel of top-40) ----
    gather_conv1<<<B_ * NPTS, 64, 0, stream>>>(x, idx40, w1, y1, gnA1, 1);
    finalize_gn<<<1, 256, 0, stream>>>(gnA1, cntL);
    conv2_wmma<<<dim3(640, B_), 256, 0, stream>>>(y1, y2, w2, g1, b1, gnA1, gnA2);
    finalize_gn<<<1, 256, 0, stream>>>(gnA2, cntL);
    pool_norm<<<B_ * NPTS, 64, 0, stream>>>(y2, g2, b2, gnA2, x1raw);
    conv1d_wmma<<<dim3(32, B_), 256, 0, stream>>>(x1raw, wc, yraw, gnAc);
    finalize_gn<<<1, 256, 0, stream>>>(gnAc, cntC);
    row_norm<<<B_ * 64, 256, 0, stream>>>(yraw, gc, bc, gnAc, x1n, cs1, 1);
    se_excite<<<B_, 64, 0, stream>>>(cs1, s1a, s1b, e1);
    se_apply<<<B_ * 64, 256, 0, stream>>>(x1n, e1, xcat, 0);

    // ---- branch 2 (top-20 = first 20 of sorted top-40) ----
    gather_conv1<<<B_ * NPTS, 64, 0, stream>>>(x, idx40, w12, y1, gnB1, 0);
    finalize_gn<<<1, 256, 0, stream>>>(gnB1, cntL);
    conv2_wmma<<<dim3(640, B_), 256, 0, stream>>>(y1, y2, w22, g12, b12, gnB1, gnB2);
    finalize_gn<<<1, 256, 0, stream>>>(gnB2, cntL);
    pool_norm<<<B_ * NPTS, 64, 0, stream>>>(y2, g22, b22, gnB2, x1raw);
    conv1d_wmma<<<dim3(32, B_), 256, 0, stream>>>(x1raw, wc2, yraw, gnBc);
    finalize_gn<<<1, 256, 0, stream>>>(gnBc, cntC);
    row_norm<<<B_ * 64, 256, 0, stream>>>(yraw, gc2, bc2, gnBc, x1n, cs2, 1);
    se_excite<<<B_, 64, 0, stream>>>(cs2, s2a, s2b, e2);
    se_apply<<<B_ * 64, 256, 0, stream>>>(x1n, e2, xcat, 64);

    // ---- final fuse conv ----
    conv1d_wmma<<<dim3(32, B_), 256, 0, stream>>>(xcat, wa, yraw, gnF);
    finalize_gn<<<1, 256, 0, stream>>>(gnF, cntC);
    row_norm<<<B_ * 64, 256, 0, stream>>>(yraw, ga, ba, gnF, (float*)d_out, nullptr, 0);
}
